// FunnelAttention_69200513074004
// MI455X (gfx1250) — matrix operand order repaired and checked by reading, not compiled, not run on hardware
//
#include <hip/hip_runtime.h>

#define B_  2
#define L_  1024
#define D_  1024
#define H_  16
#define DH_ 64
#define HD_ (H_*DH_)
#define L2_ (2*L_)

typedef __attribute__((ext_vector_type(16))) _Float16 v16h;
typedef __attribute__((ext_vector_type(8)))  float    v8f;
typedef __attribute__((ext_vector_type(4)))  unsigned int u32x4;
typedef __attribute__((ext_vector_type(8)))  int          i32x8;
typedef __attribute__((ext_vector_type(4)))  int          i32x4;

static __device__ __forceinline__ v8f wmma16(v16h a, v16h b, v8f c) {
  return __builtin_amdgcn_wmma_f32_16x16x32_f16(false, a, false, b, (short)0, c,
                                                false, false);
}

// ---------------------------------------------------------------------------
// Tensor Data Mover: 2D f16 tile -> LDS (D# packed per CDNA5 ISA 8.3/8.4)
// ---------------------------------------------------------------------------
#if defined(__has_builtin)
#if __has_builtin(__builtin_amdgcn_tensor_load_to_lds) && \
    __has_builtin(__builtin_amdgcn_s_wait_tensorcnt)
#define USE_TDM 1
#endif
#endif
#ifndef USE_TDM
#define USE_TDM 0
#endif

static __device__ __forceinline__ unsigned lds_lo(const void* p) {
  return (unsigned)(uintptr_t)p;  // low 32 bits of generic shared addr = LDS byte addr
}

#if USE_TDM
// invariant part of the descriptor (group 1): built once, reused per issue
static __device__ __forceinline__ i32x8 tdm_g1_2d(unsigned tile_x, unsigned tile_y,
                                                  unsigned long long stride_x,
                                                  unsigned tensor_x,
                                                  unsigned tensor_y) {
  i32x8 g1;
  g1[0] = (int)(1u << 16);                      // data_size=1 (2B), wg_mask=0
  g1[1] = (int)((tensor_x & 0xffffu) << 16);    // tensor_dim0[15:0] @ bits[63:48]
  g1[2] = (int)(((tensor_x >> 16) & 0xffffu) | ((tensor_y & 0xffffu) << 16));
  g1[3] = (int)(((tensor_y >> 16) & 0xffffu) | ((tile_x & 0xffffu) << 16));
  g1[4] = (int)(tile_y & 0xffffu);              // tile_dim1, tile_dim2=0
  g1[5] = (int)(unsigned)(stride_x & 0xffffffffull);
  g1[6] = (int)(unsigned)((stride_x >> 32) & 0xffffull);
  g1[7] = 0;
  return g1;
}

static __device__ __forceinline__ void tdm_issue(const void* gsrc,
                                                 unsigned lds_addr, i32x8 g1) {
  unsigned long long ga = (unsigned long long)(uintptr_t)gsrc;
  u32x4 g0;
  g0[0] = 1u;                                   // count=1 user descriptor
  g0[1] = lds_addr;
  g0[2] = (unsigned)ga;
  g0[3] = (unsigned)((ga >> 32) & 0x1ffffffu) | (2u << 30);  // addr[56:32], type=2
  i32x4 gz;
  gz[0] = 0; gz[1] = 0; gz[2] = 0; gz[3] = 0;
#if __clang_major__ >= 23
  i32x8 z8;
  for (int i = 0; i < 8; ++i) z8[i] = 0;
  __builtin_amdgcn_tensor_load_to_lds(g0, g1, gz, gz, z8, 0);
#else
  __builtin_amdgcn_tensor_load_to_lds(g0, g1, gz, gz, 0);
#endif
}
#endif

// ---------------------------------------------------------------------------
// fp32 -> fp16 converts
// ---------------------------------------------------------------------------
__global__ void cvt_f16_kernel(const float* __restrict__ in,
                               _Float16* __restrict__ out, int n) {
  int i = blockIdx.x * blockDim.x + threadIdx.x;
  if (i < n) out[i] = (_Float16)in[i];
}

// generic transpose convert: in [R][C] f32 -> out [C][R] f16
__global__ void cvt_w_t_kernel(const float* __restrict__ in,
                               _Float16* __restrict__ out, int R, int C) {
  int i = blockIdx.x * blockDim.x + threadIdx.x;
  if (i < R * C) {
    int r = i / C, c = i % C;
    out[(size_t)c * R + r] = (_Float16)in[i];
  }
}

// r_kernel [H,D,DH] f32 -> [H*DH, D] f16
__global__ void cvt_rk_t_kernel(const float* __restrict__ in,
                                _Float16* __restrict__ out) {
  int i = blockIdx.x * blockDim.x + threadIdx.x;
  if (i < H_ * D_ * DH_) {
    int c = i % DH_;
    int d = (i / DH_) % D_;
    int h = i / (DH_ * D_);
    out[((size_t)h * DH_ + c) * D_ + d] = (_Float16)in[i];
  }
}

// ---------------------------------------------------------------------------
// WMMA GEMM: C[M,N] = A[M,K] * BT[N,K]^T (+bias[n])
// One wave per 16x64 C tile. A tile TDM-DMA'd, double-buffered, last tile
// peeled so the steady-state loop is branch-free. 4 B fragments preloaded,
// then 4 independent WMMAs (one loadcnt wait per tile).
// permMode: 0 = f32 row-major (+resid), 1 = f16 [B,H,L,DH], 2 = f16 [B,H,DH,L]
// ---------------------------------------------------------------------------
__global__ __launch_bounds__(32) void gemm_wmma_kernel(
    const _Float16* __restrict__ A, const _Float16* __restrict__ BT,
    const float* __restrict__ bias,
    _Float16* __restrict__ out16, int permL, int permMode,
    float* __restrict__ out32, const float* __restrict__ resid,
    int M, int N, int K) {
  __shared__ _Float16 sA[2][16 * 32];
  const int lane  = threadIdx.x;
  const int m0    = blockIdx.y * 16;
  const int n0    = blockIdx.x * 64;
  const int mrow  = lane & 15;
  const int halfw = lane >> 4;

  v8f acc[4];
  for (int nt = 0; nt < 4; ++nt)
    for (int r = 0; r < 8; ++r) acc[nt][r] = 0.f;

  auto load_bfrag = [&](int nt, int k0) -> v16h {
    return *(const v16h*)(BT + (size_t)(n0 + nt * 16 + mrow) * K + k0 +
                          halfw * 16);
  };
  auto compute_tile = [&](const _Float16* sAc, int k0) {
    v16h af;
#pragma unroll
    for (int i = 0; i < 16; ++i) {
      int kk = (halfw << 4) + i;
      af[i] = sAc[mrow * 32 + kk];
    }
    v16h b0 = load_bfrag(0, k0);
    v16h b1 = load_bfrag(1, k0);
    v16h b2 = load_bfrag(2, k0);
    v16h b3 = load_bfrag(3, k0);
    acc[0] = wmma16(af, b0, acc[0]);
    acc[1] = wmma16(af, b1, acc[1]);
    acc[2] = wmma16(af, b2, acc[2]);
    acc[3] = wmma16(af, b3, acc[3]);
  };

#if USE_TDM
  const unsigned ldsA0 = lds_lo(&sA[0][0]);
  const unsigned ldsA1 = lds_lo(&sA[1][0]);
  const i32x8 g1A =
      tdm_g1_2d(32, 16, (unsigned long long)K, (unsigned)K, (unsigned)M);
  tdm_issue(A + (size_t)m0 * K, ldsA0, g1A);          // tile 0 -> buf 0
  int k0 = 0;
  for (; k0 < K - 32; k0 += 32) {                     // branch-free steady state
    const int cur = (k0 >> 5) & 1;
    tdm_issue(A + (size_t)m0 * K + k0 + 32, cur ? ldsA0 : ldsA1, g1A);
    __builtin_amdgcn_s_wait_tensorcnt(1);             // current tile resident
    __builtin_prefetch(BT + (size_t)n0 * K + k0 + 32, 0, 3);
    compute_tile(&sA[cur][0], k0);
  }
  __builtin_amdgcn_s_wait_tensorcnt(0);               // peeled last tile
  compute_tile(&sA[(k0 >> 5) & 1][0], k0);
#else
  for (int k0 = 0; k0 < K; k0 += 32) {
    for (int t = lane; t < 64; t += 32) {
      int r = t >> 2, c4 = t & 3;
      ((float4*)&sA[0][0])[t] =
          ((const float4*)(A + (size_t)(m0 + r) * K + k0))[c4];
    }
    __syncthreads();
    compute_tile(&sA[0][0], k0);
    __syncthreads();
  }
#endif

  // epilogue (C layout: row = halfw*8 + vgpr, col = lane&15)
#pragma unroll
  for (int nt = 0; nt < 4; ++nt)
    for (int r = 0; r < 8; ++r) {
      int m = m0 + halfw * 8 + r;
      int n = n0 + nt * 16 + mrow;
      float v = acc[nt][r];
      if (bias) v += bias[n];
      if (out16) {
        int bI = m / permL, l = m % permL, hh = n >> 6, c = n & 63;
        if (permMode == 1)
          out16[(((size_t)bI * H_ + hh) * permL + l) * DH_ + c] = (_Float16)v;
        else
          out16[(((size_t)bI * H_ + hh) * DH_ + c) * permL + l] = (_Float16)v;
      }
      if (out32) {
        if (resid) v += resid[(size_t)m * N + n];
        out32[(size_t)m * N + n] = v;
      }
    }
}

// ---------------------------------------------------------------------------
// Fused flash attention. rel_shift collapses to pos[q,k] = aR(q) . pr[L-q+k]:
// per 16x32 score tile a 16x48 WMMA band over pr rows, diagonal-gathered via
// LDS. All B fragments are contiguous 32B global loads (weights/K/pr natural,
// V consumed transposed [B,H,DH,L]). Fragments preloaded, WMMAs interleaved
// across accumulators for XDL ILP.
// ---------------------------------------------------------------------------
__global__ __launch_bounds__(32) void attn_flash_kernel(
    const _Float16* __restrict__ qh, const _Float16* __restrict__ kh,
    const _Float16* __restrict__ vhT, const _Float16* __restrict__ pr,
    const float* __restrict__ rwb, const float* __restrict__ rrb,
    const float* __restrict__ rsb, const float* __restrict__ segemb,
    const unsigned char* __restrict__ segmat,
    const unsigned char* __restrict__ mask,
    _Float16* __restrict__ attn_out) {
  __shared__ _Float16 sQ[16 * 64];
  __shared__ float    sPos[16 * 48];
  __shared__ _Float16 sP[16 * 32];
  __shared__ float    sST[16 * 2];

  const int lane  = threadIdx.x;
  const int mrow  = lane & 15;
  const int halfw = lane >> 4;
  const int qt = blockIdx.x % (L_ / 16);
  const int h  = (blockIdx.x / (L_ / 16)) % H_;
  const int b  = blockIdx.x / (L_ / 16) / H_;
  const int q0 = qt * 16;
  const float scale = 0.125f;
  const size_t bhL  = ((size_t)b * H_ + h) * L_;
  const size_t bhDH = ((size_t)b * H_ + h) * DH_;

#if USE_TDM
  tdm_issue(qh + (bhL + q0) * DH_, lds_lo(&sQ[0]),
            tdm_g1_2d(64, 16, 64ull, 64u, (unsigned)(B_ * H_ * L_)));
  __builtin_amdgcn_s_wait_tensorcnt(0);
#else
  for (int t = lane; t < 128; t += 32) {
    int r = t >> 3, c4 = t & 7;
    ((float4*)sQ)[t] = ((const float4*)(qh + (bhL + q0 + r) * DH_))[c4];
  }
#endif
  __syncthreads();

  // per-row segment logits: st_s = (qh + r_s_bias*scale) . seg_embed[s]
  if (lane < 16) {
    float s0 = 0.f, s1 = 0.f;
    for (int d = 0; d < DH_; ++d) {
      float qv = (float)sQ[lane * DH_ + d] + rsb[h * DH_ + d] * scale;
      s0 += qv * segemb[(0 * H_ + h) * DH_ + d];
      s1 += qv * segemb[(1 * H_ + h) * DH_ + d];
    }
    sST[lane * 2 + 0] = s0;
    sST[lane * 2 + 1] = s1;
  }
  __syncthreads();

  // A fragments, scale + biases folded into the q side
  v16h aW[2], aR[2];
#pragma unroll
  for (int dc = 0; dc < 2; ++dc)
    for (int i = 0; i < 16; ++i) {
      int kk = (halfw << 4) + i;
      int d  = dc * 32 + kk;
      float qv = (float)sQ[mrow * DH_ + d];
      aW[dc][i] = (_Float16)((qv + rwb[h * DH_ + d]) * scale);
      aR[dc][i] = (_Float16)((qv + rrb[h * DH_ + d]) * scale);
    }

  float mrun[8], lrun[8];
  v8f O[4];
  for (int nt = 0; nt < 4; ++nt)
    for (int r = 0; r < 8; ++r) O[nt][r] = 0.f;
  for (int r = 0; r < 8; ++r) { mrun[r] = -1e30f; lrun[r] = 0.f; }

  for (int k0 = 0; k0 < L_; k0 += 32) {
    // ---- content scores: preload 4 K fragments, then interleaved WMMAs
    v8f sc[2];
    for (int nt = 0; nt < 2; ++nt)
      for (int r = 0; r < 8; ++r) sc[nt][r] = 0.f;
    {
      const _Float16* kbase = kh + (bhL + k0 + mrow) * DH_ + halfw * 16;
      v16h k00 = *(const v16h*)(kbase);                       // nt0 dc0
      v16h k01 = *(const v16h*)(kbase + 32);                  // nt0 dc1
      v16h k10 = *(const v16h*)(kbase + 16 * DH_);            // nt1 dc0
      v16h k11 = *(const v16h*)(kbase + 16 * DH_ + 32);       // nt1 dc1
      sc[0] = wmma16(aW[0], k00, sc[0]);
      sc[1] = wmma16(aW[0], k10, sc[1]);
      sc[0] = wmma16(aW[1], k01, sc[0]);
      sc[1] = wmma16(aW[1], k11, sc[1]);
    }

    // ---- relative-position band P[16,48] over pr rows j_lo..j_lo+47
    int j_lo = L_ - q0 - 15 + k0;
    v8f pz[3];
    for (int nt = 0; nt < 3; ++nt)
      for (int r = 0; r < 8; ++r) pz[nt][r] = 0.f;
    {
      v16h bp0[3], bp1[3];
#pragma unroll
      for (int nt = 0; nt < 3; ++nt) {
        int j = j_lo + nt * 16 + mrow;
        j = (j < 0) ? 0 : ((j > L2_ - 1) ? (L2_ - 1) : j);
        const _Float16* pbase = pr + ((size_t)h * L2_ + j) * DH_ + halfw * 16;
        bp0[nt] = *(const v16h*)(pbase);
        bp1[nt] = *(const v16h*)(pbase + 32);
      }
      pz[0] = wmma16(aR[0], bp0[0], pz[0]);
      pz[1] = wmma16(aR[0], bp0[1], pz[1]);
      pz[2] = wmma16(aR[0], bp0[2], pz[2]);
      pz[0] = wmma16(aR[1], bp1[0], pz[0]);
      pz[1] = wmma16(aR[1], bp1[1], pz[1]);
      pz[2] = wmma16(aR[1], bp1[2], pz[2]);
    }
    for (int nt = 0; nt < 3; ++nt)
      for (int r = 0; r < 8; ++r)
        sPos[(halfw * 8 + r) * 48 + nt * 16 + mrow] = pz[nt][r];
    __syncthreads();

    // ---- combine pos diagonal gather + segment select + mask
    for (int nt = 0; nt < 2; ++nt)
      for (int r = 0; r < 8; ++r) {
        int qi = halfw * 8 + r;
        int ki = nt * 16 + mrow;
        float val = sc[nt][r] + sPos[qi * 48 + (15 - qi + ki)];
        val += segmat[((size_t)b * L_ + q0 + qi) * L_ + k0 + ki]
                   ? sST[qi * 2 + 1] : sST[qi * 2 + 0];
        val -= 1.0e6f * (float)mask[(size_t)b * L_ + k0 + ki];
        sc[nt][r] = val;
      }

    // ---- online softmax (rows live across 16-lane half-groups)
    float alpha[8];
    for (int r = 0; r < 8; ++r) {
      float mx = fmaxf(sc[0][r], sc[1][r]);
      for (int off = 8; off; off >>= 1) mx = fmaxf(mx, __shfl_xor(mx, off, 32));
      float mnew = fmaxf(mrun[r], mx);
      alpha[r] = __expf(mrun[r] - mnew);
      mrun[r] = mnew;
    }
    for (int r = 0; r < 8; ++r) {
      float p0 = __expf(sc[0][r] - mrun[r]);
      float p1 = __expf(sc[1][r] - mrun[r]);
      sc[0][r] = p0; sc[1][r] = p1;
      float rs = p0 + p1;
      for (int off = 8; off; off >>= 1) rs += __shfl_xor(rs, off, 32);
      lrun[r] = lrun[r] * alpha[r] + rs;
      for (int nt = 0; nt < 4; ++nt) O[nt][r] *= alpha[r];
    }

    // ---- P -> LDS (f16 row-major) for A-layout reload
    for (int nt = 0; nt < 2; ++nt)
      for (int r = 0; r < 8; ++r)
        sP[(halfw * 8 + r) * 32 + nt * 16 + mrow] = (_Float16)sc[nt][r];
    __syncthreads();

    // ---- O += P(16x32) * V(32x64); V fragments contiguous from vhT
    v16h aP;
#pragma unroll
    for (int i = 0; i < 16; ++i) {
      int kk = (halfw << 4) + i;
      aP[i] = sP[mrow * 32 + kk];
    }
    {
      const _Float16* vbase = vhT + (bhDH + mrow) * L_ + k0 + halfw * 16;
      v16h bv0 = *(const v16h*)(vbase);
      v16h bv1 = *(const v16h*)(vbase + 16 * L_);
      v16h bv2 = *(const v16h*)(vbase + 32 * L_);
      v16h bv3 = *(const v16h*)(vbase + 48 * L_);
      O[0] = wmma16(aP, bv0, O[0]);
      O[1] = wmma16(aP, bv1, O[1]);
      O[2] = wmma16(aP, bv2, O[2]);
      O[3] = wmma16(aP, bv3, O[3]);
    }
    __syncthreads();
  }

  // normalize and write [B, L, H*DH] so out-proj is a plain GEMM
  for (int nt = 0; nt < 4; ++nt)
    for (int r = 0; r < 8; ++r) {
      int qi = halfw * 8 + r;
      int n  = nt * 16 + mrow;
      float v = O[nt][r] / lrun[r];
      attn_out[((size_t)b * L_ + q0 + qi) * HD_ + h * DH_ + n] = (_Float16)v;
    }
}

// ---------------------------------------------------------------------------
// LayerNorm over D=1024 per row
// ---------------------------------------------------------------------------
__global__ __launch_bounds__(256) void ln_kernel(const float* __restrict__ x,
                                                 const float* __restrict__ gamma,
                                                 const float* __restrict__ beta,
                                                 float* __restrict__ out) {
  __shared__ float sbuf[16];
  const int row = blockIdx.x;
  float s = 0.f, s2 = 0.f;
  for (int i = threadIdx.x; i < D_; i += 256) {
    float v = x[(size_t)row * D_ + i];
    s += v; s2 += v * v;
  }
  for (int off = 16; off; off >>= 1) {
    s  += __shfl_xor(s, off, 32);
    s2 += __shfl_xor(s2, off, 32);
  }
  int w = threadIdx.x >> 5;
  if ((threadIdx.x & 31) == 0) { sbuf[w] = s; sbuf[8 + w] = s2; }
  __syncthreads();
  if (threadIdx.x == 0) {
    float a = 0.f, c = 0.f;
    for (int i = 0; i < 8; ++i) { a += sbuf[i]; c += sbuf[8 + i]; }
    sbuf[0] = a; sbuf[8] = c;
  }
  __syncthreads();
  float mu  = sbuf[0] / (float)D_;
  float var = sbuf[8] / (float)D_ - mu * mu;
  float inv = rsqrtf(var + 1e-5f);
  for (int i = threadIdx.x; i < D_; i += 256)
    out[(size_t)row * D_ + i] =
        (x[(size_t)row * D_ + i] - mu) * inv * gamma[i] + beta[i];
}

// ---------------------------------------------------------------------------
extern "C" void kernel_launch(void* const* d_in, const int* in_sizes, int n_in,
                              void* d_out, int out_size, void* d_ws,
                              size_t ws_size, hipStream_t stream) {
  (void)in_sizes; (void)n_in; (void)out_size; (void)ws_size;
  const float* q    = (const float*)d_in[0];
  const float* k    = (const float*)d_in[1];
  const float* v    = (const float*)d_in[2];
  const unsigned char* mask   = (const unsigned char*)d_in[3];
  const float* pos  = (const float*)d_in[4];
  const unsigned char* segmat = (const unsigned char*)d_in[5];
  const float* Wq   = (const float*)d_in[6];
  const float* Wk   = (const float*)d_in[7];
  const float* bk   = (const float*)d_in[8];
  const float* Wv   = (const float*)d_in[9];
  const float* bv   = (const float*)d_in[10];
  const float* rwb  = (const float*)d_in[11];
  const float* rrb  = (const float*)d_in[12];
  const float* rker = (const float*)d_in[13];
  const float* rsb  = (const float*)d_in[14];
  const float* seg  = (const float*)d_in[15];
  const float* Wo   = (const float*)d_in[16];
  const float* bo   = (const float*)d_in[17];
  const float* gamma= (const float*)d_in[18];
  const float* beta = (const float*)d_in[19];
  float* out = (float*)d_out;

  char* w = (char*)d_ws;
  size_t off = 0;
  auto carve = [&](size_t bytes) -> void* {
    void* p = w + off;
    off += (bytes + 255) & ~(size_t)255;
    return p;
  };
  const size_t BLD = (size_t)B_ * L_ * D_;
  const size_t WSZ = (size_t)D_ * HD_;
  _Float16* q16   = (_Float16*)carve(BLD * 2);
  _Float16* k16   = (_Float16*)carve(BLD * 2);
  _Float16* v16   = (_Float16*)carve(BLD * 2);
  _Float16* WqT   = (_Float16*)carve(WSZ * 2);
  _Float16* WkT   = (_Float16*)carve(WSZ * 2);
  _Float16* WvT   = (_Float16*)carve(WSZ * 2);
  _Float16* WoT   = (_Float16*)carve(WSZ * 2);
  _Float16* rkT   = (_Float16*)carve(WSZ * 2);
  _Float16* pos16 = (_Float16*)carve((size_t)L2_ * D_ * 2);
  _Float16* qh16  = (_Float16*)carve((size_t)B_ * H_ * L_ * DH_ * 2);
  _Float16* kh16  = (_Float16*)carve((size_t)B_ * H_ * L_ * DH_ * 2);
  _Float16* vhT16 = (_Float16*)carve((size_t)B_ * H_ * L_ * DH_ * 2);
  _Float16* pr16  = (_Float16*)carve((size_t)H_ * L2_ * DH_ * 2);
  _Float16* at16  = (_Float16*)carve((size_t)B_ * L_ * HD_ * 2);
  float*    x32   = (float*)carve(BLD * 4);

  const int T = 256;
  auto nb = [&](size_t n) { return (unsigned)((n + T - 1) / T); };

  cvt_f16_kernel<<<nb(BLD), T, 0, stream>>>(q, q16, (int)BLD);
  cvt_f16_kernel<<<nb(BLD), T, 0, stream>>>(k, k16, (int)BLD);
  cvt_f16_kernel<<<nb(BLD), T, 0, stream>>>(v, v16, (int)BLD);
  cvt_f16_kernel<<<nb((size_t)L2_ * D_), T, 0, stream>>>(
      pos, pos16, (int)((size_t)L2_ * D_));
  cvt_w_t_kernel<<<nb(WSZ), T, 0, stream>>>(Wq, WqT, D_, HD_);
  cvt_w_t_kernel<<<nb(WSZ), T, 0, stream>>>(Wk, WkT, D_, HD_);
  cvt_w_t_kernel<<<nb(WSZ), T, 0, stream>>>(Wv, WvT, D_, HD_);
  cvt_w_t_kernel<<<nb(WSZ), T, 0, stream>>>(Wo, WoT, HD_, D_);
  cvt_rk_t_kernel<<<nb(WSZ), T, 0, stream>>>(rker, rkT);

  dim3 g16(HD_ / 64, (B_ * L_) / 16);
  gemm_wmma_kernel<<<g16, 32, 0, stream>>>(q16, WqT, nullptr, qh16, L_, 1,
                                           nullptr, nullptr, B_ * L_, HD_, D_);
  gemm_wmma_kernel<<<g16, 32, 0, stream>>>(k16, WkT, bk, kh16, L_, 1,
                                           nullptr, nullptr, B_ * L_, HD_, D_);
  gemm_wmma_kernel<<<g16, 32, 0, stream>>>(v16, WvT, bv, vhT16, L_, 2,
                                           nullptr, nullptr, B_ * L_, HD_, D_);
  gemm_wmma_kernel<<<g16, 32, 0, stream>>>(pos16, rkT, nullptr, pr16, L2_, 1,
                                           nullptr, nullptr, L2_, HD_, D_);

  attn_flash_kernel<<<B_ * H_ * (L_ / 16), 32, 0, stream>>>(
      qh16, kh16, vhT16, pr16, rwb, rrb, rsb, seg, segmat, mask, at16);

  gemm_wmma_kernel<<<g16, 32, 0, stream>>>(at16, WoT, bo, nullptr, 1, 0,
                                           x32, q, B_ * L_, D_, HD_);

  ln_kernel<<<B_ * L_, 256, 0, stream>>>(x32, gamma, beta, out);
}
